// LiteMLA_23854248362823
// MI455X (gfx1250) — compile-verified, matrix operand order checked by reading, not run on hardware
//
#include <hip/hip_runtime.h>
#include <hip/hip_bf16.h>

typedef __attribute__((ext_vector_type(16))) __bf16 v16bf;
typedef __attribute__((ext_vector_type(8)))  __bf16 v8bf;
typedef __attribute__((ext_vector_type(4)))  __bf16 v4bf;
typedef __attribute__((ext_vector_type(2)))  __bf16 v2bf;
typedef __attribute__((ext_vector_type(8)))  float  v8f;

#define BATCH   8
#define HW      4096
#define CIN     256
#define C3      768     // 3*TOTAL
#define CATT    512     // 2*TOTAL
#define COUT    256

// ---- workspace layout (bytes) ----
#define OFF_WQ   ((size_t)0)                       // 768*256 bf16  = 393216
#define OFF_WP   ((size_t)393216)                  // 256*512 bf16  = 262144
#define OFF_SC   ((size_t)655360)                  // 256 f32
#define OFF_SH   ((size_t)656384)                  // 256 f32
#define OFF_QKV  ((size_t)1048576)                 // 8*768*4096 bf16 = 50331648
#define OFF_DW   (OFF_QKV + (size_t)50331648)      // dw scratch (reused for att)
#define OFF_AGG  (OFF_DW  + (size_t)50331648)
#define OFF_ATT  OFF_DW                            // att out: 8*512*4096 bf16 = 33.5MB

// load 4 consecutive elements as f32 (vectorized: b128 for f32 src, b64 for bf16 src)
__device__ __forceinline__ void ld4_f32(const float* p, float o[4]) {
    float4 v = *(const float4*)p;
    o[0] = v.x; o[1] = v.y; o[2] = v.z; o[3] = v.w;
}
__device__ __forceinline__ void ld4_f32(const __bf16* p, float o[4]) {
    v4bf v = *(const v4bf*)p;
    o[0] = (float)v[0]; o[1] = (float)v[1]; o[2] = (float)v[2]; o[3] = (float)v[3];
}

// ---------------- prep: weight conversion + BN folding ----------------
__global__ __launch_bounds__(256) void prep_kernel(
    const float* __restrict__ wqkv, const float* __restrict__ wproj,
    const float* __restrict__ gma,  const float* __restrict__ bta,
    const float* __restrict__ mean, const float* __restrict__ var,
    __bf16* __restrict__ wq_b, __bf16* __restrict__ wp_b,
    float* __restrict__ scale, float* __restrict__ shift)
{
    int i = blockIdx.x * blockDim.x + threadIdx.x;
    int stride = gridDim.x * blockDim.x;
    for (int j = i; j < C3 * CIN; j += stride)   wq_b[j] = (__bf16)wqkv[j];
    for (int j = i; j < COUT * CATT; j += stride) wp_b[j] = (__bf16)wproj[j];
    if (i < COUT) {
        float inv = gma[i] * rsqrtf(var[i] + 1e-5f);
        scale[i] = inv;
        shift[i] = bta[i] - mean[i] * inv;
    }
}

// ---------------- tiled bf16 WMMA GEMM: OUT[b, M, 4096] = W[M,K] x X[b,K,4096] ----------------
// Workgroup tile: 64(M) x 128(N), 8 waves (4 M x 2 N), 4 WMMA accs per wave.
// Double-buffered LDS, software-pipelined staging (issue global loads before WMMA block).
template<typename XT, bool PROJ>
__global__ __launch_bounds__(256) void gemm_wmma_bf16(
    const __bf16* __restrict__ W, const XT* __restrict__ X,
    void* __restrict__ OUT, const float* __restrict__ scale,
    const float* __restrict__ shift, int M, int K)
{
    __shared__ __bf16 BsA[128 * 48];   // X tile, N-major, stride 48 (16B-aligned frag reads)
    __shared__ __bf16 BsB[128 * 48];

    const int tid  = threadIdx.x;
    const int lane = tid & 31;
    const int wid  = tid >> 5;
    const int wm   = wid >> 1;        // 0..3
    const int wn   = wid & 1;         // 0..1
    const int b    = blockIdx.z;
    const int m0   = blockIdx.y * 64;
    const int n0   = blockIdx.x * 128;

    const XT* Xb = X + (size_t)b * K * HW;

    // per-thread staging coordinates: 2 chunks, each = (k,k+1) x 4 consecutive n
    const int kp0 = tid >> 5;                 // 0..7
    const int n40 = (tid & 31) * 4;           // 0..124
    const int kp1 = kp0 + 8;                  // 8..15
    const int n41 = n40;

    v8f acc[4];
#pragma unroll
    for (int s = 0; s < 4; ++s)
#pragma unroll
        for (int r = 0; r < 8; ++r) acc[s][r] = 0.f;

    const int KB = K >> 5;
    const int kh = (lane >> 4) * 8;   // 0 or 8: K-half select per WMMA fragment layout

    float lo0[4], hi0[4], lo1[4], hi1[4];

    // prologue: stage K-slab 0 into BsA
    {
        const XT* p0 = Xb + (size_t)(kp0 * 2) * HW + n0 + n40;
        const XT* p1 = Xb + (size_t)(kp1 * 2) * HW + n0 + n41;
        ld4_f32(p0, lo0); ld4_f32(p0 + HW, hi0);
        ld4_f32(p1, lo1); ld4_f32(p1 + HW, hi1);
#pragma unroll
        for (int j = 0; j < 4; ++j) {
            v2bf a; a[0] = (__bf16)lo0[j]; a[1] = (__bf16)hi0[j];
            *(v2bf*)(&BsA[(n40 + j) * 48 + kp0 * 2]) = a;
            v2bf c; c[0] = (__bf16)lo1[j]; c[1] = (__bf16)hi1[j];
            *(v2bf*)(&BsA[(n41 + j) * 48 + kp1 * 2]) = c;
        }
    }
    __syncthreads();

    for (int kb = 0; kb < KB; ++kb) {
        __bf16* cur = (kb & 1) ? BsB : BsA;
        __bf16* nxt = (kb & 1) ? BsA : BsB;

        // issue next K-slab's global loads early (waits land after the WMMA block)
        if (kb + 1 < KB) {
            const XT* p0 = Xb + (size_t)((kb + 1) * 32 + kp0 * 2) * HW + n0 + n40;
            const XT* p1 = Xb + (size_t)((kb + 1) * 32 + kp1 * 2) * HW + n0 + n41;
            ld4_f32(p0, lo0); ld4_f32(p0 + HW, hi0);
            ld4_f32(p1, lo1); ld4_f32(p1 + HW, hi1);
        }

        // A fragment straight from L2-resident bf16 weights
        const __bf16* ap = W + (size_t)(m0 + wm * 16 + (lane & 15)) * K + kb * 32 + kh;
        v8bf alo = *(const v8bf*)ap;
        v8bf ahi = *(const v8bf*)(ap + 16);
        v16bf af;
#pragma unroll
        for (int i = 0; i < 8; ++i) { af[i] = alo[i]; af[i + 8] = ahi[i]; }

#pragma unroll
        for (int s = 0; s < 4; ++s) {
            const __bf16* bp = &cur[(wn * 64 + s * 16 + (lane & 15)) * 48 + kh];
            v8bf blo = *(const v8bf*)bp;          // ds_load_b128
            v8bf bhi = *(const v8bf*)(bp + 16);   // ds_load_b128
            v16bf bfv;
#pragma unroll
            for (int i = 0; i < 8; ++i) { bfv[i] = blo[i]; bfv[i + 8] = bhi[i]; }
            acc[s] = __builtin_amdgcn_wmma_f32_16x16x32_bf16(
                false, af, false, bfv, (short)0, acc[s], false, false);
        }

        // commit next slab to the other buffer (cvt_pk + packed ds_store_b32)
        if (kb + 1 < KB) {
#pragma unroll
            for (int j = 0; j < 4; ++j) {
                v2bf a; a[0] = (__bf16)lo0[j]; a[1] = (__bf16)hi0[j];
                *(v2bf*)(&nxt[(n40 + j) * 48 + kp0 * 2]) = a;
                v2bf c; c[0] = (__bf16)lo1[j]; c[1] = (__bf16)hi1[j];
                *(v2bf*)(&nxt[(n41 + j) * 48 + kp1 * 2]) = c;
            }
        }
        __syncthreads();
    }

    // epilogue: C tile layout -> VGPR r: M=r (lanes 0-15) / M=r+8 (lanes 16-31), N=lane&15
    const int mb = m0 + wm * 16 + ((lane >> 4) ? 8 : 0);
#pragma unroll
    for (int s = 0; s < 4; ++s) {
        int nc = n0 + wn * 64 + s * 16 + (lane & 15);
#pragma unroll
        for (int r = 0; r < 8; ++r) {
            int oc  = mb + r;
            float v = acc[s][r];
            if constexpr (PROJ) {
                float* Of = (float*)OUT;
                Of[((size_t)b * M + oc) * HW + nc] = v * scale[oc] + shift[oc];
            } else {
                __bf16* Ob = (__bf16*)OUT;
                Ob[((size_t)b * M + oc) * HW + nc] = (__bf16)v;
            }
        }
    }
}

// ---------------- depthwise 5x5, pad 2 ----------------
__global__ __launch_bounds__(256) void dwconv5_kernel(
    const __bf16* __restrict__ in, const float* __restrict__ wdw,
    __bf16* __restrict__ out)
{
    const int c = blockIdx.y, b = blockIdx.z;
    const int p = blockIdx.x * 256 + threadIdx.x;
    const int h = p >> 6, w = p & 63;
    const __bf16* ib = in + ((size_t)b * C3 + c) * HW;
    const float*  wb = wdw + c * 25;          // scalar-uniform per block
    float a = 0.f;
#pragma unroll
    for (int dy = 0; dy < 5; ++dy) {
        int hh = h + dy - 2;
        if ((unsigned)hh < 64u) {
#pragma unroll
            for (int dx = 0; dx < 5; ++dx) {
                int ww = w + dx - 2;
                if ((unsigned)ww < 64u)
                    a += wb[dy * 5 + dx] * (float)ib[(hh << 6) + ww];
            }
        }
    }
    out[((size_t)b * C3 + c) * HW + p] = (__bf16)a;
}

// ---------------- grouped 1x1 conv (8 -> 8 per group) ----------------
__global__ __launch_bounds__(256) void pwgroup_kernel(
    const __bf16* __restrict__ in, const float* __restrict__ wpw,
    __bf16* __restrict__ out)
{
    const int oc = blockIdx.y, b = blockIdx.z;
    const int p  = blockIdx.x * 256 + threadIdx.x;
    const int gb = (oc >> 3) << 3;
    const float* wb = wpw + oc * 8;
    const __bf16* ib = in + ((size_t)b * C3 + gb) * HW + p;
    float a = 0.f;
#pragma unroll
    for (int i = 0; i < 8; ++i) a += wb[i] * (float)ib[(size_t)i * HW];
    out[((size_t)b * C3 + oc) * HW + p] = (__bf16)a;
}

// ---------------- ReLU linear attention: one workgroup per (b, g) ----------------
__global__ __launch_bounds__(256) void attn_kernel(
    const __bf16* __restrict__ qkv, const __bf16* __restrict__ agg,
    __bf16* __restrict__ att)
{
    const int bg = blockIdx.x;
    const int b = bg >> 6, g = bg & 63;
    const __bf16* src; int cg;
    if (g < 32) { src = qkv; cg = g; } else { src = agg; cg = g - 32; }
    const __bf16* base = src + ((size_t)b * C3 + cg * 24) * HW;  // q:0-7 k:8-15 v:16-23

    const int tid = threadIdx.x;
    float acc[72];
#pragma unroll
    for (int j = 0; j < 72; ++j) acc[j] = 0.f;

    // phase 1: vk[d][e] = sum_n v_ext[d,n] * relu(k[e,n])
    for (int n = tid; n < HW; n += 256) {
        float kv[8], vv[9];
#pragma unroll
        for (int e = 0; e < 8; ++e) kv[e] = fmaxf((float)base[(size_t)(8 + e) * HW + n], 0.f);
#pragma unroll
        for (int d = 0; d < 8; ++d) vv[d] = (float)base[(size_t)(16 + d) * HW + n];
        vv[8] = 1.f;
#pragma unroll
        for (int d = 0; d < 9; ++d)
#pragma unroll
            for (int e = 0; e < 8; ++e) acc[d * 8 + e] += vv[d] * kv[e];
    }

    // wave reduce + cross-wave LDS reduce
#pragma unroll
    for (int j = 0; j < 72; ++j) {
        acc[j] += __shfl_xor(acc[j], 16, 32);
        acc[j] += __shfl_xor(acc[j],  8, 32);
        acc[j] += __shfl_xor(acc[j],  4, 32);
        acc[j] += __shfl_xor(acc[j],  2, 32);
        acc[j] += __shfl_xor(acc[j],  1, 32);
    }
    __shared__ float red[8][72];
    const int wid = tid >> 5, lane = tid & 31;
    if (lane == 0) {
#pragma unroll
        for (int j = 0; j < 72; ++j) red[wid][j] = acc[j];
    }
    __syncthreads();
    float vk[72];
#pragma unroll
    for (int j = 0; j < 72; ++j) {
        float s = 0.f;
#pragma unroll
        for (int w = 0; w < 8; ++w) s += red[w][j];
        vk[j] = s;
    }

    // phase 2: out[d,n] = (vk[d]·q[n]) / (vk[8]·q[n] + eps)
    __bf16* obase = att + ((size_t)b * CATT + g * 8) * HW;
    for (int n = tid; n < HW; n += 256) {
        float q[8];
#pragma unroll
        for (int e = 0; e < 8; ++e) q[e] = fmaxf((float)base[(size_t)e * HW + n], 0.f);
        float den = 1e-15f;
#pragma unroll
        for (int e = 0; e < 8; ++e) den += vk[64 + e] * q[e];
        float rden = 1.f / den;
#pragma unroll
        for (int d = 0; d < 8; ++d) {
            float o = 0.f;
#pragma unroll
            for (int e = 0; e < 8; ++e) o += vk[d * 8 + e] * q[e];
            obase[(size_t)d * HW + n] = (__bf16)(o * rden);
        }
    }
}

extern "C" void kernel_launch(void* const* d_in, const int* in_sizes, int n_in,
                              void* d_out, int out_size, void* d_ws, size_t ws_size,
                              hipStream_t stream) {
    const float* x     = (const float*)d_in[0];
    const float* wqkv  = (const float*)d_in[1];
    const float* wdw   = (const float*)d_in[2];
    const float* wpw   = (const float*)d_in[3];
    const float* wproj = (const float*)d_in[4];
    const float* gma   = (const float*)d_in[5];
    const float* bta   = (const float*)d_in[6];
    const float* mean  = (const float*)d_in[7];
    const float* var   = (const float*)d_in[8];

    char* ws = (char*)d_ws;
    __bf16* wq_b   = (__bf16*)(ws + OFF_WQ);
    __bf16* wp_b   = (__bf16*)(ws + OFF_WP);
    float*  scale  = (float*)(ws + OFF_SC);
    float*  shift  = (float*)(ws + OFF_SH);
    __bf16* qkv_ws = (__bf16*)(ws + OFF_QKV);
    __bf16* dw_ws  = (__bf16*)(ws + OFF_DW);
    __bf16* agg_ws = (__bf16*)(ws + OFF_AGG);
    __bf16* att_ws = (__bf16*)(ws + OFF_ATT);   // reuses dw region (ordered on stream)

    prep_kernel<<<1024, 256, 0, stream>>>(wqkv, wproj, gma, bta, mean, var,
                                          wq_b, wp_b, scale, shift);

    // qkv = W_qkv(768x256) x X
    gemm_wmma_bf16<float, false><<<dim3(HW / 128, C3 / 64, BATCH), 256, 0, stream>>>(
        wq_b, x, (void*)qkv_ws, nullptr, nullptr, C3, CIN);

    dwconv5_kernel<<<dim3(HW / 256, C3, BATCH), 256, 0, stream>>>(qkv_ws, wdw, dw_ws);
    pwgroup_kernel<<<dim3(HW / 256, C3, BATCH), 256, 0, stream>>>(dw_ws, wpw, agg_ws);

    attn_kernel<<<BATCH * 64, 256, 0, stream>>>(qkv_ws, agg_ws, att_ws);

    // out = BN( W_proj(256x512) x att )
    gemm_wmma_bf16<__bf16, true><<<dim3(HW / 128, COUT / 64, BATCH), 256, 0, stream>>>(
        wp_b, att_ws, d_out, scale, shift, COUT, CATT);
}
